// patch_attention_75625784148189
// MI455X (gfx1250) — compile-verified, matrix operand order checked by reading, not run on hardware
//
#include <hip/hip_runtime.h>

// ---------------------------------------------------------------------------
// Types for CDNA5 WMMA (wave32, 16x16x32 bf16 -> f32)
// ---------------------------------------------------------------------------
typedef __attribute__((ext_vector_type(16))) __bf16 v16bf;
typedef __attribute__((ext_vector_type(8)))  __bf16 v8bf;
typedef __attribute__((ext_vector_type(8)))  float  v8f;

// Problem constants (B=2, C=64, H=W=256, ROWS=4, HEADS=2)
#define NP   64        // P = H/ROWS
#define NN   1024      // tokens per window
#define HD   32        // head dim

__device__ __forceinline__ v16bf cat8(v8bf lo, v8bf hi) {
  return __builtin_shufflevector(lo, hi, 0, 1, 2, 3, 4, 5, 6, 7,
                                 8, 9, 10, 11, 12, 13, 14, 15);
}

__device__ __forceinline__ v8bf ld8(const __bf16* p) {
  return *(const v8bf*)p;                 // one 16-byte load (global or DS)
}

__device__ __forceinline__ v8f bf16mma(v16bf a, v16bf b, v8f c) {
  return __builtin_amdgcn_wmma_f32_16x16x32_bf16(false, a, false, b,
                                                 (short)0, c, false, false);
}

// 16x16 output tile of  feat(16x64) @ Wrow(bf16, row-major 64) : 2 WMMAs
__device__ __forceinline__ v8f projb(v16bf a0, v16bf a1,
                                     const __bf16* __restrict__ Wrow,
                                     unsigned bse) {
  v16bf b0 = cat8(ld8(Wrow + bse),      ld8(Wrow + 16 + bse));
  v16bf b1 = cat8(ld8(Wrow + 32 + bse), ld8(Wrow + 48 + bse));
  v8f acc = {};
  acc = bf16mma(a0, b0, acc);
  acc = bf16mma(a1, b1, acc);
  return acc;
}

// ---------------------------------------------------------------------------
// Kernel 0: one-shot f32 -> bf16 conversion of the four 64x64 weights.
// 4 blocks (one per matrix) x 256 threads x 16 elements.
// ---------------------------------------------------------------------------
__global__ void __launch_bounds__(256) k_wcvt(
    const float* __restrict__ Wq, const float* __restrict__ Wk,
    const float* __restrict__ Wv, const float* __restrict__ Wp,
    __bf16* __restrict__ wb) {
  const float* srcs[4] = {Wq, Wk, Wv, Wp};
  const float* src = srcs[blockIdx.x];
  __bf16* dst = wb + blockIdx.x * 4096;
  int i0 = threadIdx.x * 16;
#pragma unroll
  for (int i = 0; i < 16; i++) dst[i0 + i] = (__bf16)src[i0 + i];
}

// ---------------------------------------------------------------------------
// Kernel 1: q/k/v projections, x[B,C,H,W] f32 -> q,k (bf16 [bph][n][d]),
//           vT (bf16 [bph][d][n]).  Scale 1/sqrt(32) folded into q.
// One wave per 16-token tile: A-operand gathered ONCE, all 4 out-channel
// tiles and all 3 matrices computed from it (24 WMMAs / wave).
// ---------------------------------------------------------------------------
__global__ void __launch_bounds__(256) k_qkv(
    const float* __restrict__ x, const __bf16* __restrict__ wb,
    __bf16* __restrict__ q, __bf16* __restrict__ k, __bf16* __restrict__ vT) {
  const int lane = threadIdx.x & 31;
  const int tokTile = blockIdx.x * 8 + (threadIdx.x >> 5);  // 8192 tiles
  const unsigned bse = (lane >> 4) * 8;

  // A operands: token rows of feat, channels gathered (stride H*W) from x
  int T = tokTile * 16 + (lane & 15);
  int b = T >> 16, rem = T & 65535;
  int p = rem >> 10, n = rem & 1023;
  int h = p * 4 + (n >> 8), w = n & 255;
  const float* xp = x + ((unsigned long long)b << 22) + (unsigned)(h * 256 + w);
  v8bf t0, t1, t2, t3;
#pragma unroll
  for (int i = 0; i < 8; i++) {
    t0[i] = (__bf16)xp[(unsigned long long)(bse + i) << 16];
    t1[i] = (__bf16)xp[(unsigned long long)(16 + bse + i) << 16];
    t2[i] = (__bf16)xp[(unsigned long long)(32 + bse + i) << 16];
    t3[i] = (__bf16)xp[(unsigned long long)(48 + bse + i) << 16];
  }
  v16bf a0 = cat8(t0, t1);
  v16bf a1 = cat8(t2, t3);

  const __bf16* wq = wb;
  const __bf16* wk = wb + 4096;
  const __bf16* wv = wb + 8192;

#pragma unroll
  for (int ot = 0; ot < 4; ot++) {
    const int o = ot * 16 + (lane & 15);        // out channel for this lane
    v8f accq = projb(a0, a1, wq + o * 64, bse);
    v8f acck = projb(a0, a1, wk + o * 64, bse);
    v8f accv = projb(a0, a1, wv + o * 64, bse);

    const int head = o >> 5, d = o & 31;
#pragma unroll
    for (int r = 0; r < 8; r++) {
      int Tr = tokTile * 16 + r + 8 * (lane >> 4);
      int bb = Tr >> 16, rem2 = Tr & 65535;
      int pp = rem2 >> 10, nn = rem2 & 1023;
      unsigned long long bphb =
          (unsigned long long)(((bb * NP + pp) * 2 + head)) * 32768ull;
      q[bphb + (unsigned long long)nn * HD + d] =
          (__bf16)(accq[r] * 0.17677669529663687f);
      k[bphb + (unsigned long long)nn * HD + d] = (__bf16)acck[r];
      vT[bphb + (unsigned long long)d * NN + nn] = (__bf16)accv[r];
    }
  }
}

// ---------------------------------------------------------------------------
// Kernel 2: windowed attention, flash-style online softmax.
// Block = one (b,p,head, 128-row q slab); 8 waves x 16 q rows.
// K (64KB) + V^T (64KB) async-staged into LDS, P-tile bounced via LDS scratch.
// Row-sums of exp(P) computed by an extra WMMA against an all-ones B matrix.
// ---------------------------------------------------------------------------
__global__ void __launch_bounds__(256) k_attn(
    const __bf16* __restrict__ q, const __bf16* __restrict__ k,
    const __bf16* __restrict__ vT, __bf16* __restrict__ ao) {
  extern __shared__ char smem[];
  __bf16* sK = (__bf16*)smem;          // [1024 keys][32 dims]
  __bf16* sV = sK + 32768;             // [32 dims][1024 keys]
  __bf16* sP = sV + 32768;             // 8 waves * [16][32]

  const int tid = threadIdx.x, lane = tid & 31, wave = tid >> 5;
  const int s = blockIdx.x;
  const int qblk = s & 7;              // which 128-row slab
  const int bph = s >> 3;              // (b*P + p)*2 + head
  const unsigned long long base = (unsigned long long)bph * 32768ull;
  const __bf16* kp = k + base;
  const __bf16* vp = vT + base;

  // ---- async fill of K and V^T into LDS (CDNA5 async-to-LDS path) ----
  {
    unsigned ldsK = (unsigned)(unsigned long long)sK;
    unsigned ldsV = (unsigned)(unsigned long long)sV;
#pragma unroll
    for (int i = 0; i < 16; i++) {
      unsigned off = (unsigned)(i * 256 + tid) * 16u;
      asm volatile("global_load_async_to_lds_b128 %0, %1, %2"
                   :: "v"(ldsK + off), "v"(off), "s"(kp) : "memory");
      asm volatile("global_load_async_to_lds_b128 %0, %1, %2"
                   :: "v"(ldsV + off), "v"(off), "s"(vp) : "memory");
    }
    asm volatile("s_wait_asynccnt 0" ::: "memory");
  }
  __syncthreads();

  // ---- per-wave Q tile (16 rows x 32 dims) in WMMA A layout ----
  const int qrow0 = qblk * 128 + wave * 16;
  const unsigned bse = (lane >> 4) * 8;
  const __bf16* qp = q + base + (unsigned long long)(qrow0 + (lane & 15)) * HD;
  v16bf qa = cat8(ld8(qp + bse), ld8(qp + 16 + bse));

  v16bf onesB;                         // all-ones B operand for row sums
#pragma unroll
  for (int i = 0; i < 16; i++) onesB[i] = (__bf16)1.0f;

  float m[8], l[8];
  v8f O0 = {}, O1 = {};
#pragma unroll
  for (int r = 0; r < 8; r++) { m[r] = -1e30f; l[r] = 0.f; }
  __bf16* sPw = sP + wave * 512;

  for (int j = 0; j < 32; j++) {       // 32 chunks of 32 keys
    // K^T B-operands from LDS
    const __bf16* kr0 = sK + (unsigned)(j * 32 + (lane & 15)) * HD;
    const __bf16* kr1 = kr0 + 16 * HD;
    v16bf kb0 = cat8(ld8(kr0 + bse), ld8(kr0 + 16 + bse));
    v16bf kb1 = cat8(ld8(kr1 + bse), ld8(kr1 + 16 + bse));

    v8f Z = {};
    v8f S0 = bf16mma(qa, kb0, Z);      // scores, scale pre-folded into q
    v8f S1 = bf16mma(qa, kb1, Z);

    // row max of this chunk (reduce across 16 lanes of each half)
    float mc[8];
#pragma unroll
    for (int r = 0; r < 8; r++) mc[r] = fmaxf(S0[r], S1[r]);
#pragma unroll
    for (int msk = 1; msk <= 8; msk <<= 1)
#pragma unroll
      for (int r = 0; r < 8; r++)
        mc[r] = fmaxf(mc[r], __shfl_xor(mc[r], msk, 32));

    float alpha[8];
#pragma unroll
    for (int r = 0; r < 8; r++) {
      float mn = fmaxf(m[r], mc[r]);
      alpha[r] = __expf(m[r] - mn);
      m[r] = mn;
    }
#pragma unroll
    for (int r = 0; r < 8; r++) {
      S0[r] = __expf(S0[r] - m[r]);
      S1[r] = __expf(S1[r] - m[r]);
    }

    // D-layout -> A-layout for P via per-wave LDS scratch (DS in-order)
    {
      const int Mh = 8 * (lane >> 4), n0 = lane & 15;
#pragma unroll
      for (int r = 0; r < 8; r++) {
        sPw[(r + Mh) * 32 + n0]      = (__bf16)S0[r];
        sPw[(r + Mh) * 32 + 16 + n0] = (__bf16)S1[r];
      }
    }
    __builtin_amdgcn_wave_barrier();
    const __bf16* pr = sPw + (lane & 15) * 32;
    v16bf pa = cat8(ld8(pr + bse), ld8(pr + 16 + bse));

    // row sums of exp(P) via WMMA against ones (replaces 32 lane shuffles)
    v8f rsv = bf16mma(pa, onesB, Z);
#pragma unroll
    for (int r = 0; r < 8; r++) {
      l[r] = l[r] * alpha[r] + rsv[r];
      O0[r] *= alpha[r];
      O1[r] *= alpha[r];
    }

    // V B-operands from transposed LDS rows (contiguous)
    const __bf16* vr0 = sV + (unsigned)(lane & 15) * NN + j * 32;
    const __bf16* vr1 = vr0 + 16 * NN;
    v16bf vb0 = cat8(ld8(vr0 + bse), ld8(vr0 + 16 + bse));
    v16bf vb1 = cat8(ld8(vr1 + bse), ld8(vr1 + 16 + bse));

    O0 = bf16mma(pa, vb0, O0);
    O1 = bf16mma(pa, vb1, O1);
  }

  // epilogue: normalize, store attention output token-major bf16 [bp][n][64]
  const unsigned long long aob = (unsigned long long)(bph >> 1) * 65536ull;
  const int head = bph & 1;
#pragma unroll
  for (int r = 0; r < 8; r++) {
    float inv = 1.0f / l[r];
    int n = qrow0 + r + 8 * (lane >> 4);
    unsigned long long rowb = aob + (unsigned long long)n * 64 + head * 32;
    ao[rowb + (lane & 15)]      = (__bf16)(O0[r] * inv);
    ao[rowb + 16 + (lane & 15)] = (__bf16)(O1[r] * inv);
  }
}

// ---------------------------------------------------------------------------
// Kernel 3: output projection  ao[T][64] @ Wp^T -> out[B,C,H,W] f32
// One wave per 16-token tile; A loaded once, 4 out-channel tiles computed.
// ---------------------------------------------------------------------------
__global__ void __launch_bounds__(256) k_out(
    const __bf16* __restrict__ ao, const __bf16* __restrict__ wb,
    float* __restrict__ out) {
  const int lane = threadIdx.x & 31;
  const int tokTile = blockIdx.x * 8 + (threadIdx.x >> 5);
  const unsigned bse = (lane >> 4) * 8;

  const __bf16* row =
      ao + (unsigned long long)(tokTile * 16 + (lane & 15)) * 64;
  v16bf a0 = cat8(ld8(row + bse),      ld8(row + 16 + bse));
  v16bf a1 = cat8(ld8(row + 32 + bse), ld8(row + 48 + bse));

  const __bf16* wp = wb + 12288;       // Wp in bf16 weight block

#pragma unroll
  for (int ot = 0; ot < 4; ot++) {
    const int o = ot * 16 + (lane & 15);
    v8f acc = projb(a0, a1, wp + o * 64, bse);

#pragma unroll
    for (int r = 0; r < 8; r++) {
      int Tr = tokTile * 16 + r + 8 * (lane >> 4);
      int bb = Tr >> 16, rem = Tr & 65535;
      int pp = rem >> 10, nn = rem & 1023;
      int hh = pp * 4 + (nn >> 8), ww = nn & 255;
      out[(((unsigned long long)bb * 64 + o) << 16) +
          (unsigned)(hh * 256 + ww)] = acc[r];
    }
  }
}

// ---------------------------------------------------------------------------
// Host launch.  Workspace: q | k | vT | ao (16 MB each, bf16) | wb (32 KB).
// ---------------------------------------------------------------------------
extern "C" void kernel_launch(void* const* d_in, const int* in_sizes, int n_in,
                              void* d_out, int out_size, void* d_ws,
                              size_t ws_size, hipStream_t stream) {
  const float* x  = (const float*)d_in[0];
  const float* Wq = (const float*)d_in[1];
  const float* Wk = (const float*)d_in[2];
  const float* Wv = (const float*)d_in[3];
  const float* Wp = (const float*)d_in[4];
  float* out = (float*)d_out;

  char* ws = (char*)d_ws;
  __bf16* q  = (__bf16*)(ws);
  __bf16* k  = (__bf16*)(ws + 16ull * 1024 * 1024);
  __bf16* vT = (__bf16*)(ws + 32ull * 1024 * 1024);
  __bf16* ao = (__bf16*)(ws + 48ull * 1024 * 1024);
  __bf16* wb = (__bf16*)(ws + 64ull * 1024 * 1024);   // 4 x 4096 bf16

  // one-shot weight conversion (32 KB, stays L2-resident)
  k_wcvt<<<4, 256, 0, stream>>>(Wq, Wk, Wv, Wp, wb);
  // 8192 token-tiles, 8 waves/block, each wave does all 4 out-channel tiles
  k_qkv<<<1024, 256, 0, stream>>>(x, wb, q, k, vT);
  // 256 (b,p,head) windows x 8 q-slabs; 136 KB dynamic LDS
  k_attn<<<2048, 256, 139264, stream>>>(q, k, vT, ao);
  k_out<<<1024, 256, 0, stream>>>(ao, wb, out);
}
